// AliceAttention_61160334295239
// MI455X (gfx1250) — matrix operand order repaired and checked by reading, not compiled, not run on hardware
//
#include <hip/hip_runtime.h>
#include <hip/hip_bf16.h>
#include <math.h>

typedef __bf16 bf16_t;
typedef __attribute__((ext_vector_type(16))) __bf16 v16bf;
typedef __attribute__((ext_vector_type(8)))  __bf16 v8bf;
typedef __attribute__((ext_vector_type(8)))  float  v8f;
typedef __attribute__((ext_vector_type(4)))  int    v4i;

#define B_   2
#define S_   2048
#define H_   4096
#define NH_  32
#define HD_  128
#define BS_  (B_*S_)          // 4096 rows
#define NELEM_ (B_*S_*H_)     // 16777216 (== H*H as well)

#define LDS_AS __attribute__((address_space(3)))
#define GLB_AS __attribute__((address_space(1)))

union Frag16 { v16bf v; v8bf h[2]; };

// ---------------------------------------------------------------------------
// CDNA5 async global->LDS copy (ASYNCcnt-tracked, no VGPR staging).
// Builtin signature (from toolchain diagnostic): (v4i addrspace(1)*,
// v4i addrspace(3)*, int offset, int cpol).
// ---------------------------------------------------------------------------
__device__ __forceinline__ void async_cp_b128(const bf16_t* g, bf16_t* l) {
  unsigned int loff = (unsigned int)(uintptr_t)l;
#if __has_builtin(__builtin_amdgcn_global_load_async_to_lds_b128)
  __builtin_amdgcn_global_load_async_to_lds_b128(
      (GLB_AS v4i*)(uintptr_t)g, (LDS_AS v4i*)loff, 0, 0);
#else
  unsigned long long ga = (unsigned long long)(uintptr_t)g;
  asm volatile("global_load_async_to_lds_b128 %0, %1, off"
               :: "v"(loff), "v"(ga) : "memory");
#endif
}

__device__ __forceinline__ void wait_async0() {
#if __has_builtin(__builtin_amdgcn_s_wait_asynccnt)
  __builtin_amdgcn_s_wait_asynccnt(0);
#else
  asm volatile("s_wait_asynccnt 0x0" ::: "memory");
#endif
}

// ---------------------------------------------------------------------------
// fp32 -> bf16 convert (grid-stride)
// ---------------------------------------------------------------------------
__global__ void cvt_f32_bf16(const float* __restrict__ in, bf16_t* __restrict__ out, int n) {
  int i = blockIdx.x * blockDim.x + threadIdx.x;
  int stride = gridDim.x * blockDim.x;
  for (; i < n; i += stride) out[i] = (bf16_t)in[i];
}

// ---------------------------------------------------------------------------
// C = A(MxK bf16 rm) * Bt(NxK bf16 rm)^T, fp32 accumulate.
// 128 threads / 4 waves; block tile 128x128; wave tile 64x64 (4x4 WMMA accs).
// Double-buffered LDS tiles (128x32 A + 128x32 B, padded rows) filled with
// async global->LDS copies: prefetch K-slice k+32 while WMMAs consume slice k
// from LDS; s_wait_asynccnt + barrier at the iteration boundary.
// OUTMODE: 0 = bf16 MxN, 1 = fp32 MxN, 2 = bf16 transposed to (B,NH,HD,S).
// ---------------------------------------------------------------------------
template <int OUTMODE>
__global__ __launch_bounds__(128)
void gemm_bf16(const bf16_t* __restrict__ A, const bf16_t* __restrict__ Bt,
               void* __restrict__ Cout, int M, int N, int K) {
  __shared__ __align__(16) bf16_t Alds[2][128 * 40];
  __shared__ __align__(16) bf16_t Blds[2][128 * 40];

  const int tid  = threadIdx.x;
  const int wave = tid >> 5;
  const int lane = tid & 31;
  const int ln   = lane & 15;
  const int half = lane >> 4;
  const int m_base = blockIdx.y * 128;
  const int n_base = blockIdx.x * 128;
  const int mw = (wave >> 1) * 64;
  const int nw = (wave & 1) * 64;

  auto stage = [&](int buf, int k0) {
    #pragma unroll
    for (int i = 0; i < 4; i++) {
      int cid = tid + 128 * i;           // 0..511 : 128 rows x 4 16B-chunks
      int row = cid >> 2;
      int c4  = cid & 3;
      async_cp_b128(A  + (size_t)(m_base + row) * K + k0 + c4 * 8,
                    &Alds[buf][row * 40 + c4 * 8]);
      async_cp_b128(Bt + (size_t)(n_base + row) * K + k0 + c4 * 8,
                    &Blds[buf][row * 40 + c4 * 8]);
    }
  };

  const v8f vzero = {0.f,0.f,0.f,0.f,0.f,0.f,0.f,0.f};
  v8f acc[4][4];
  #pragma unroll
  for (int i = 0; i < 4; i++)
    #pragma unroll
    for (int j = 0; j < 4; j++) acc[i][j] = vzero;

  stage(0, 0);
  wait_async0();
  __syncthreads();

  int cur = 0;
  for (int k0 = 0; k0 < K; k0 += 32) {
    if (k0 + 32 < K) stage(cur ^ 1, k0 + 32);   // async prefetch next K-slice

    Frag16 a[4], b[4];
    #pragma unroll
    for (int mi = 0; mi < 4; mi++) {
      const bf16_t* ap = &Alds[cur][(mw + mi * 16 + ln) * 40 + half * 16];
      a[mi].h[0] = *(const v8bf*)(ap);
      a[mi].h[1] = *(const v8bf*)(ap + 8);
    }
    #pragma unroll
    for (int ni = 0; ni < 4; ni++) {
      const bf16_t* bp = &Blds[cur][(nw + ni * 16 + ln) * 40 + half * 16];
      b[ni].h[0] = *(const v8bf*)(bp);
      b[ni].h[1] = *(const v8bf*)(bp + 8);
    }
    #pragma unroll
    for (int mi = 0; mi < 4; mi++)
      #pragma unroll
      for (int ni = 0; ni < 4; ni++)
        acc[mi][ni] = __builtin_amdgcn_wmma_f32_16x16x32_bf16(
            false, a[mi].v, false, b[ni].v, (short)0, acc[mi][ni], false, false);

    wait_async0();
    __syncthreads();
    cur ^= 1;
  }

  #pragma unroll
  for (int mi = 0; mi < 4; mi++)
    #pragma unroll
    for (int ni = 0; ni < 4; ni++)
      #pragma unroll
      for (int g = 0; g < 8; g++) {
        int row = m_base + mw + mi * 16 + g + 8 * half;
        int col = n_base + nw + ni * 16 + ln;
        float v = acc[mi][ni][g];
        if (OUTMODE == 0) {
          ((bf16_t*)Cout)[(size_t)row * N + col] = (bf16_t)v;
        } else if (OUTMODE == 1) {
          ((float*)Cout)[(size_t)row * N + col] = v;
        } else {  // V output pre-transposed to (B, NH, HD, S) for attention
          int bb = row >> 11, s = row & (S_ - 1);
          int head = col >> 7, d = col & (HD_ - 1);
          ((bf16_t*)Cout)[(((size_t)bb * NH_ + head) * HD_ + d) * S_ + s] = (bf16_t)v;
        }
      }
}

// ---------------------------------------------------------------------------
// In-place RoPE on bf16 Q and K, layout (B,S,NH,HD). Bijective on (d, d+64).
// ---------------------------------------------------------------------------
__global__ void rope_inplace(bf16_t* __restrict__ Q, bf16_t* __restrict__ Kv, int n) {
  int i = blockIdx.x * blockDim.x + threadIdx.x;
  int stride = gridDim.x * blockDim.x;
  const float nl = -logf(10000.f) / 64.f;
  for (; i < n; i += stride) {
    int d  = i & 63;
    int nh = (i >> 6) & (NH_ - 1);
    int s  = (i >> 11) & (S_ - 1);
    int b  = i >> 22;
    size_t base = ((size_t)(b * S_ + s)) * H_ + (size_t)nh * HD_;
    float ang = (float)s * expf(nl * (float)d);
    float c = cosf(ang), sn = sinf(ang);
    float q0 = (float)Q[base + d], q1 = (float)Q[base + d + 64];
    Q[base + d]      = (bf16_t)(q0 * c - q1 * sn);
    Q[base + d + 64] = (bf16_t)(q1 * c + q0 * sn);
    float k0 = (float)Kv[base + d], k1 = (float)Kv[base + d + 64];
    Kv[base + d]      = (bf16_t)(k0 * c - k1 * sn);
    Kv[base + d + 64] = (bf16_t)(k1 * c + k0 * sn);
  }
}

// ---------------------------------------------------------------------------
// Flash-style causal attention. 1 wave = 16 query rows of one (b, head).
// Q (B,S,NH,HD), K (B,S,NH,HD), Vt (B,NH,HD,S) -- all bf16.
// Per 32-kv block: 8 K fragments loaded up front, then 8 chained WMMAs;
// online softmax with 16-lane shfl_xor reductions; P goes C-layout ->
// A-layout via a wave-private LDS tile; P*V uses contiguous Vt fragments.
// ---------------------------------------------------------------------------
__global__ __launch_bounds__(128)
void attn_fwd(const bf16_t* __restrict__ Q, const bf16_t* __restrict__ K,
              const bf16_t* __restrict__ Vt, bf16_t* __restrict__ O) {
  __shared__ __align__(32) bf16_t Plds[4][16][40];

  const int wave = threadIdx.x >> 5;
  const int lane = threadIdx.x & 31;
  const int ln   = lane & 15;
  const int half = lane >> 4;

  int qb   = blockIdx.x * 4 + wave;        // 0 .. B*NH*(S/16)-1
  int b    = qb / (NH_ * (S_ / 16));
  int rem  = qb % (NH_ * (S_ / 16));
  int head = rem / (S_ / 16);
  int q0   = (rem % (S_ / 16)) * 16;

  const size_t headoff = (size_t)b * S_ * H_ + (size_t)head * HD_;      // Q/K/O
  const bf16_t* vhead  = Vt + (size_t)(b * NH_ + head) * HD_ * S_;      // Vt

  // Q fragments (live in registers across the whole kv loop)
  v16bf qa[4];
  {
    const bf16_t* qrow = Q + headoff + (size_t)(q0 + ln) * H_;
    #pragma unroll
    for (int kc = 0; kc < 4; kc++) {
      Frag16 f;
      f.h[0] = *(const v8bf*)(qrow + kc * 32 + half * 16);
      f.h[1] = *(const v8bf*)(qrow + kc * 32 + half * 16 + 8);
      qa[kc] = f.v;
    }
  }

  const v8f vzero = {0.f,0.f,0.f,0.f,0.f,0.f,0.f,0.f};
  v8f acc_o[8];
  #pragma unroll
  for (int i = 0; i < 8; i++) acc_o[i] = vzero;
  float mrow[8], lrow[8];
  #pragma unroll
  for (int g = 0; g < 8; g++) { mrow[g] = -3.0e38f; lrow[g] = 0.f; }

  const float rsf = 0.08838834764831845f;  // 1/sqrt(128)

  for (int kv0 = 0; kv0 < q0 + 16; kv0 += 32) {
    // ---- load ALL K fragments for this 32-kv block, then run the WMMAs
    v16bf kb[2][4];
    #pragma unroll
    for (int t = 0; t < 2; t++) {
      const bf16_t* krow = K + headoff + (size_t)(kv0 + t * 16 + ln) * H_ + half * 16;
      #pragma unroll
      for (int kc = 0; kc < 4; kc++) kb[t][kc] = *(const v16bf*)(krow + kc * 32);
    }
    v8f sacc[2];
    sacc[0] = vzero; sacc[1] = vzero;
    #pragma unroll
    for (int t = 0; t < 2; t++)
      #pragma unroll
      for (int kc = 0; kc < 4; kc++)
        sacc[t] = __builtin_amdgcn_wmma_f32_16x16x32_bf16(
            false, qa[kc], false, kb[t][kc], (short)0, sacc[t], false, false);

    // ---- online softmax update + write P to LDS (bf16)
    #pragma unroll
    for (int g = 0; g < 8; g++) {
      int row = q0 + g + 8 * half;
      int c0 = kv0 + ln, c1 = kv0 + 16 + ln;
      float s0 = sacc[0][g] * rsf + ((c0 > row) ? -1.0e9f : 0.f);
      float s1 = sacc[1][g] * rsf + ((c1 > row) ? -1.0e9f : 0.f);
      float tm = fmaxf(s0, s1);
      #pragma unroll
      for (int off = 1; off < 16; off <<= 1) tm = fmaxf(tm, __shfl_xor(tm, off, 32));
      float mnew  = fmaxf(mrow[g], tm);
      float scale = __expf(mrow[g] - mnew);
      float p0 = __expf(s0 - mnew);
      float p1 = __expf(s1 - mnew);
      float rs = p0 + p1;
      #pragma unroll
      for (int off = 1; off < 16; off <<= 1) rs += __shfl_xor(rs, off, 32);
      lrow[g] = lrow[g] * scale + rs;
      mrow[g] = mnew;
      #pragma unroll
      for (int ni = 0; ni < 8; ni++) acc_o[ni][g] *= scale;
      Plds[wave][g + 8 * half][ln]      = (bf16_t)p0;
      Plds[wave][g + 8 * half][ln + 16] = (bf16_t)p1;
    }
    __builtin_amdgcn_wave_barrier();   // LDS in-order per wave; stop reordering

    // ---- P (16x32, A-layout from LDS) x V (32x128) from pre-transposed Vt
    Frag16 pa;
    pa.h[0] = *(const v8bf*)(&Plds[wave][ln][half * 16]);
    pa.h[1] = *(const v8bf*)(&Plds[wave][ln][half * 16 + 8]);
    __builtin_amdgcn_wave_barrier();
    #pragma unroll
    for (int ni = 0; ni < 8; ni++) {
      const bf16_t* vp = vhead + (size_t)(ni * 16 + ln) * S_ + kv0 + half * 16;
      v16bf vb = *(const v16bf*)vp;    // 16 contiguous kv values for column d
      acc_o[ni] = __builtin_amdgcn_wmma_f32_16x16x32_bf16(
          false, pa.v, false, vb, (short)0, acc_o[ni], false, false);
    }
  }

  // ---- epilogue: divide by row sums, store bf16 attention output (B,S,H)
  #pragma unroll
  for (int ni = 0; ni < 8; ni++)
    #pragma unroll
    for (int g = 0; g < 8; g++) {
      int row = q0 + g + 8 * half;
      float inv = 1.0f / lrow[g];
      O[headoff + (size_t)row * H_ + ni * 16 + ln] = (bf16_t)(acc_o[ni][g] * inv);
    }
}

// ---------------------------------------------------------------------------
// Launch
// ---------------------------------------------------------------------------
extern "C" void kernel_launch(void* const* d_in, const int* in_sizes, int n_in,
                              void* d_out, int out_size, void* d_ws, size_t ws_size,
                              hipStream_t stream) {
  (void)in_sizes; (void)n_in; (void)out_size; (void)ws_size;
  const float* hs = (const float*)d_in[0];
  // d_in[1] = attention_mask: causal -1e9 mask, applied analytically in-kernel.
  const float* Wq = (const float*)d_in[2];
  const float* Wk = (const float*)d_in[3];
  const float* Wv = (const float*)d_in[4];
  const float* Wo = (const float*)d_in[5];

  const size_t NB = (size_t)NELEM_ * sizeof(bf16_t);  // 32 MB per bf16 tensor
  char* ws = (char*)d_ws;
  bf16_t* Xb  = (bf16_t*)(ws + 0 * NB);
  bf16_t* Wqb = (bf16_t*)(ws + 1 * NB);
  bf16_t* Wkb = (bf16_t*)(ws + 2 * NB);
  bf16_t* Wvb = (bf16_t*)(ws + 3 * NB);
  bf16_t* Wob = (bf16_t*)(ws + 4 * NB);
  bf16_t* Qb  = (bf16_t*)(ws + 5 * NB);
  bf16_t* Kb  = (bf16_t*)(ws + 6 * NB);
  bf16_t* Vtb = (bf16_t*)(ws + 7 * NB);   // (B,NH,HD,S)
  bf16_t* Ab  = (bf16_t*)(ws + 8 * NB);

  // 1) fp32 -> bf16 staging (everything then fits in the 192MB L2)
  cvt_f32_bf16<<<2048, 256, 0, stream>>>(hs, Xb,  NELEM_);
  cvt_f32_bf16<<<2048, 256, 0, stream>>>(Wq, Wqb, NELEM_);
  cvt_f32_bf16<<<2048, 256, 0, stream>>>(Wk, Wkb, NELEM_);
  cvt_f32_bf16<<<2048, 256, 0, stream>>>(Wv, Wvb, NELEM_);
  cvt_f32_bf16<<<2048, 256, 0, stream>>>(Wo, Wob, NELEM_);

  // 2) QKV projections (V written pre-transposed for the attention kernel)
  dim3 ggrid(H_ / 128, BS_ / 128);
  gemm_bf16<0><<<ggrid, 128, 0, stream>>>(Xb, Wqb, (void*)Qb,  BS_, H_, H_);
  gemm_bf16<0><<<ggrid, 128, 0, stream>>>(Xb, Wkb, (void*)Kb,  BS_, H_, H_);
  gemm_bf16<2><<<ggrid, 128, 0, stream>>>(Xb, Wvb, (void*)Vtb, BS_, H_, H_);

  // 3) RoPE in place on Q and K
  rope_inplace<<<4096, 256, 0, stream>>>(Qb, Kb, B_ * S_ * NH_ * 64);

  // 4) Flash causal attention -> Ab (B,S,H) bf16
  attn_fwd<<<(B_ * NH_ * (S_ / 16)) / 4, 128, 0, stream>>>(Qb, Kb, Vtb, Ab);

  // 5) Output projection: out = A * Wo^T, fp32 result to d_out
  gemm_bf16<1><<<ggrid, 128, 0, stream>>>(Ab, Wob, d_out, BS_, H_, H_);
}
